// ModulatedStyleConv_33921651704088
// MI455X (gfx1250) — matrix operand order repaired and checked by reading, not compiled, not run on hardware
//
#include <hip/hip_runtime.h>
#include <hip/hip_bf16.h>

typedef _Float16 v8h  __attribute__((ext_vector_type(8)));
typedef _Float16 v16h __attribute__((ext_vector_type(16)));
typedef float    v8f  __attribute__((ext_vector_type(8)));
typedef unsigned int v4u __attribute__((ext_vector_type(4)));
typedef int      v8i  __attribute__((ext_vector_type(8)));
typedef int      v4i  __attribute__((ext_vector_type(4)));

#define NB    16
#define CIN   512
#define COUT  512
#define STY   512
#define HIMG  64
#define WIMG  64
#define KTAPS 9

// ---------------------------------------------------------------------------
// Kernel 1: style modulation  s[n][ci] = style[n] . style_w[ci] + style_b[ci]
// ---------------------------------------------------------------------------
__global__ __launch_bounds__(256)
void style_kernel(const float* __restrict__ style,
                  const float* __restrict__ style_w,
                  const float* __restrict__ style_b,
                  float* __restrict__ s)
{
    int idx = blockIdx.x * 256 + threadIdx.x;      // 0 .. N*CIN-1
    int n  = idx >> 9;
    int ci = idx & (CIN - 1);
    float acc = style_b[ci];
    const float* st = style + (size_t)n * STY;
    const float* sw = style_w + (size_t)ci * STY;
    #pragma unroll 8
    for (int k = 0; k < STY; ++k) acc += st[k] * sw[k];
    s[idx] = acc;
}

// ---------------------------------------------------------------------------
// Kernel 2: x (N,CIN,H,W) fp32  ->  xcl[n][h][w][ci] f16 (channels-last)
// ---------------------------------------------------------------------------
__global__ __launch_bounds__(256)
void pack_x_kernel(const float* __restrict__ x, _Float16* __restrict__ xcl)
{
    size_t idx = (size_t)blockIdx.x * 256 + threadIdx.x;   // over N*CIN*H*W
    int w = (int)(idx & (WIMG - 1));
    size_t r = idx >> 6;
    int h = (int)(r & (HIMG - 1)); r >>= 6;
    int ci = (int)(r & (CIN - 1)); r >>= 9;
    int n = (int)r;
    size_t dst = ((((size_t)n * HIMG + h) * WIMG) + w) * CIN + ci;
    xcl[dst] = (_Float16)x[idx];
}

// ---------------------------------------------------------------------------
// Kernel 3: per-(n,co) modulate + demodulate, emit f16 weights laid out as
// wf16[n][tap][co][ci]
// ---------------------------------------------------------------------------
__global__ __launch_bounds__(256)
void modw_kernel(const float* __restrict__ weight,   // (COUT, CIN, 3, 3)
                 const float* __restrict__ s,        // (N, CIN)
                 _Float16* __restrict__ wf16)
{
    __shared__ float red[256];
    const int co = blockIdx.x;
    const int n  = blockIdx.y;
    const int tid = threadIdx.x;

    const float* wrow = weight + (size_t)co * CIN * KTAPS;
    const float* srow = s + (size_t)n * CIN;

    float sumsq = 0.0f;
    for (int e = tid; e < CIN * KTAPS; e += 256) {
        int ci = e / KTAPS;
        int t  = e - ci * KTAPS;
        float v = wrow[(size_t)ci * KTAPS + t] * srow[ci];
        sumsq += v * v;
    }
    red[tid] = sumsq;
    __syncthreads();
    #pragma unroll
    for (int off = 128; off > 0; off >>= 1) {
        if (tid < off) red[tid] += red[tid + off];
        __syncthreads();
    }
    const float demod = rsqrtf(red[0] + 1e-8f);

    for (int e = tid; e < CIN * KTAPS; e += 256) {
        int ci = e / KTAPS;
        int t  = e - ci * KTAPS;
        float v = wrow[(size_t)ci * KTAPS + t] * srow[ci] * demod;
        wf16[(((size_t)(n * KTAPS + t) * COUT + co) * CIN) + ci] = (_Float16)v;
    }
}

// ---------------------------------------------------------------------------
// TDM helper: 2-D tile load (2-byte elements) via Tensor Data Mover.
// D# per CDNA5 ISA sect. 8.3/8.4.
// ---------------------------------------------------------------------------
__device__ __forceinline__ unsigned lds_off_of(const void* p)
{
    return (unsigned)(unsigned long long)(size_t)p;
}

__device__ __forceinline__ void tdm_load_2d(unsigned lds_off, const void* gptr,
                                            unsigned tensor_d0, unsigned tensor_d1,
                                            unsigned tile_d0,   unsigned tile_d1,
                                            unsigned stride_d0)
{
    unsigned long long ga = (unsigned long long)(size_t)gptr;
    v4u g0;
    g0.x = 1u;                                         // count=1, user mode
    g0.y = lds_off;                                    // lds_addr
    g0.z = (unsigned)(ga & 0xffffffffu);               // global_addr[31:0]
    g0.w = (unsigned)((ga >> 32) & 0x01ffffffu)        // global_addr[56:32]
         | (2u << 30);                                 // type=2 (image)
    v8i g1;
    g1[0] = (int)(1u << 16);                           // data_size=2 bytes
    g1[1] = (int)((tensor_d0 & 0xffffu) << 16);        // tensor_dim0[15:0]
    g1[2] = (int)((tensor_d0 >> 16) | ((tensor_d1 & 0xffffu) << 16));
    g1[3] = (int)((tensor_d1 >> 16) | (tile_d0 << 16));// tile_dim0
    g1[4] = (int)(tile_d1 & 0xffffu);                  // tile_dim1 (tile_dim2=0)
    g1[5] = (int)stride_d0;                            // tensor_dim0_stride[31:0]
    g1[6] = 0;                                         // stride[47:32], d1stride[15:0]
    g1[7] = 0;                                         // d1stride[47:16]
    v4i z4 = {};
#if __clang_major__ >= 23
    v8i z8 = {};
    __builtin_amdgcn_tensor_load_to_lds(g0, g1, z4, z4, z8, 0);
#else
    __builtin_amdgcn_tensor_load_to_lds(g0, g1, z4, z4, 0);
#endif
}

// ---------------------------------------------------------------------------
// Kernel 4: implicit-GEMM conv via v_wmma_f32_16x16x32_f16 with DOUBLE-
// BUFFERED TDM staging.  Block: 128 thr = 4 waves (2 M-waves x 2 N-waves),
// tile 32 co x 64 pixels (one image row at height h).
// Pipeline: per ci-chunk c, wave 0 issues chunk c+1's TDM descriptors into
// the other buffer, then s_wait_tensorcnt <= (#ops of c+1): TENSORcnt is
// in-order per wave, so this guarantees chunk c has landed while chunk c+1
// is still in flight behind the WMMAs.
// LDS: A 2x9x32x32 f16 (36 KB) + B 2x3x66x32 f16 (25.3 KB) = 62.2 KB.
// ---------------------------------------------------------------------------
__global__ __launch_bounds__(128)
void conv_wmma_kernel(const _Float16* __restrict__ wf16,   // [n][tap][co][ci]
                      const _Float16* __restrict__ xcl,    // [n][h][w][ci]
                      const float* __restrict__ noise,     // (N,1,H,W)
                      const float* __restrict__ nw,        // (1,)
                      float* __restrict__ out)             // (N,COUT,H,W)
{
    __shared__ _Float16 ALds[2][KTAPS][32][32];  // per-tap A tiles, 2 buffers
    __shared__ _Float16 BLds[2][3][66][32];      // halo strip, col c <-> w=c-1

    const int h     = blockIdx.x;        // 0..63
    const int coblk = blockIdx.y;        // 0..15
    const int n     = blockIdx.z;        // 0..15

    const int tid   = threadIdx.x;
    const int wave  = tid >> 5;
    const int lane  = tid & 31;
    const int mwave = wave >> 1;         // 0..1  (16 co rows each)
    const int nwave = wave & 1;          // 0..1  (32 pixels each)
    const int lhalf = lane >> 4;         // 0/1
    const int l16   = lane & 15;

    // A fragment (16x32 f16): lanes 0-15: M=lane, K {0..15};
    //                         lanes 16-31: M=lane-16, K {16..31}
    const int mrow = mwave * 16 + l16;
    const int kbA  = lhalf ? 16 : 0;
    // B fragment (32x16 f16): lane holds column N=l16; K contiguous
    const int kbB  = lhalf ? 16 : 0;
    const int pix0 = nwave * 32 + l16;
    const int pix1 = nwave * 32 + 16 + l16;

    // Zero both halo strips once: TDM only rewrites interior cols 1..64 of
    // in-image rows, so padding columns / out-of-image rows stay zero.
    for (int e = tid; e < 2 * 3 * 66 * 32; e += 128)
        ((_Float16*)BLds)[e] = (_Float16)0.0f;

    const bool row0ok = (h > 0);
    const bool row2ok = (h < HIMG - 1);
    const bool full   = row0ok && row2ok;    // 12 TDM ops per chunk, else 11

    const _Float16* aG0 = wf16 + ((size_t)(n * KTAPS) * COUT + coblk * 32) * CIN;
    const _Float16* bG0 = xcl + ((size_t)n * HIMG) * WIMG * CIN;

    v8f acc0 = {};
    v8f acc1 = {};

    // ----- stage chunk 0 into buffer 0 -----
    if (wave == 0) {
        #pragma unroll
        for (int t = 0; t < KTAPS; ++t)
            tdm_load_2d(lds_off_of(&ALds[0][t][0][0]),
                        aG0 + (size_t)t * COUT * CIN, CIN, COUT, 32, 32, CIN);
        #pragma unroll
        for (int r = 0; r < 3; ++r) {
            if ((r == 0 && !row0ok) || (r == 2 && !row2ok)) continue;
            tdm_load_2d(lds_off_of(&BLds[0][r][1][0]),
                        bG0 + ((size_t)(h + r - 1) * WIMG) * CIN,
                        CIN, WIMG, 32, 64, CIN);
        }
    }

    for (int cb = 0; cb < CIN; cb += 32) {
        const int cur = (cb >> 5) & 1;
        const int nxt = cur ^ 1;
        const bool hasnext = (cb + 32) < CIN;

        if (wave == 0) {
            if (hasnext) {
                const int cb2 = cb + 32;
                #pragma unroll
                for (int t = 0; t < KTAPS; ++t)
                    tdm_load_2d(lds_off_of(&ALds[nxt][t][0][0]),
                                aG0 + (size_t)t * COUT * CIN + cb2,
                                CIN, COUT, 32, 32, CIN);
                #pragma unroll
                for (int r = 0; r < 3; ++r) {
                    if ((r == 0 && !row0ok) || (r == 2 && !row2ok)) continue;
                    tdm_load_2d(lds_off_of(&BLds[nxt][r][1][0]),
                                bG0 + ((size_t)(h + r - 1) * WIMG) * CIN + cb2,
                                CIN, WIMG, 32, 64, CIN);
                }
                // in-order TENSORcnt: <= (#ops of chunk cb+32) => chunk cb done
                if (full) __builtin_amdgcn_s_wait_tensorcnt(12);
                else      __builtin_amdgcn_s_wait_tensorcnt(11);
            } else {
                __builtin_amdgcn_s_wait_tensorcnt(0);
            }
        }
        __syncthreads();   // chunk cb's LDS tiles visible to all waves

        #pragma unroll
        for (int t = 0; t < KTAPS; ++t) {
            const int dh  = t / 3;      // strip row
            const int dwp = t % 3;      // col offset (w + dw maps to col pix+dwp)

            v8h alo = *(const v8h*)&ALds[cur][t][mrow][kbA];
            v8h ahi = *(const v8h*)&ALds[cur][t][mrow][kbA + 8];
            v16h afrag = __builtin_shufflevector(alo, ahi,
                0, 1, 2, 3, 4, 5, 6, 7, 8, 9, 10, 11, 12, 13, 14, 15);

            v16h b0 = *(const v16h*)&BLds[cur][dh][pix0 + dwp][kbB];
            v16h b1 = *(const v16h*)&BLds[cur][dh][pix1 + dwp][kbB];

            acc0 = __builtin_amdgcn_wmma_f32_16x16x32_f16(
                false, afrag, false, b0, (short)0, acc0, false, false);
            acc1 = __builtin_amdgcn_wmma_f32_16x16x32_f16(
                false, afrag, false, b1, (short)0, acc1, false, false);
        }
        __syncthreads();   // compute(cb) done -> buffer may be re-filled
    }

    // Epilogue: noise injection + LeakyReLU(0.2). C/D layout (ISA 7.12.2):
    // lanes 0-15: VGPR r -> M=r, N=l16 ; lanes 16-31: VGPR r -> M=8+r, N=l16
    const float nwv = nw[0];
    const float nz0 = nwv * noise[((size_t)n * HIMG + h) * WIMG + pix0];
    const float nz1 = nwv * noise[((size_t)n * HIMG + h) * WIMG + pix1];
    const int coBase = coblk * 32 + mwave * 16 + (lhalf ? 8 : 0);

    #pragma unroll
    for (int r = 0; r < 8; ++r) {
        const int co = coBase + r;
        float v0 = acc0[r] + nz0;
        float v1 = acc1[r] + nz1;
        v0 = (v0 >= 0.0f) ? v0 : 0.2f * v0;
        v1 = (v1 >= 0.0f) ? v1 : 0.2f * v1;
        size_t base = (((size_t)n * COUT + co) * HIMG + h) * WIMG;
        out[base + pix0] = v0;
        out[base + pix1] = v1;
    }
}

// ---------------------------------------------------------------------------
// Launch
// ---------------------------------------------------------------------------
extern "C" void kernel_launch(void* const* d_in, const int* in_sizes, int n_in,
                              void* d_out, int out_size, void* d_ws, size_t ws_size,
                              hipStream_t stream)
{
    const float* x       = (const float*)d_in[0];   // (16,512,64,64)
    const float* style   = (const float*)d_in[1];   // (16,512)
    const float* noise   = (const float*)d_in[2];   // (16,1,64,64)
    const float* weight  = (const float*)d_in[3];   // (1,512,512,3,3)
    const float* style_w = (const float*)d_in[4];   // (512,512)
    const float* style_b = (const float*)d_in[5];   // (512,)
    const float* nw      = (const float*)d_in[6];   // (1,)
    float* out = (float*)d_out;

    char* ws = (char*)d_ws;
    float*    s    = (float*)ws;                                   // 32 KB
    _Float16* wf16 = (_Float16*)(ws + 65536);                      // 72 MB
    _Float16* xcl  = (_Float16*)(ws + 65536 +
                         (size_t)NB * KTAPS * COUT * CIN * 2);     // 64 MB

    style_kernel<<<(NB * CIN) / 256, 256, 0, stream>>>(style, style_w, style_b, s);
    pack_x_kernel<<<(NB * CIN * HIMG * WIMG) / 256, 256, 0, stream>>>(x, xcl);
    modw_kernel<<<dim3(COUT, NB), 256, 0, stream>>>(weight, s, wf16);
    conv_wmma_kernel<<<dim3(HIMG, COUT / 32, NB), 128, 0, stream>>>(
        wf16, xcl, noise, nw, out);
}
